// DPMultiHeadAttention_17068200034325
// MI455X (gfx1250) — matrix operand order repaired and checked by reading, not compiled, not run on hardware
//
#include <hip/hip_runtime.h>

typedef __attribute__((ext_vector_type(16))) __bf16 v16bf;
typedef __attribute__((ext_vector_type(8)))  float  v8f;
typedef unsigned int uint;

constexpr int Bb = 2, Sq = 2048, Dm = 1024, Hn = 16, Dh = 64;

__device__ __forceinline__ v8f wmma_bf16(v16bf a, v16bf b, v8f c) {
  return __builtin_amdgcn_wmma_f32_16x16x32_bf16(false, a, false, b, (short)0, c,
                                                 false, false);
}

// Issue one 16-byte async global->LDS copy per lane (GVS mode, ASYNCcnt).
__device__ __forceinline__ void async_ld128(uint lds_addr, uint goff,
                                            const void* gbase) {
  asm volatile("global_load_async_to_lds_b128 %0, %1, %2"
               :: "v"(lds_addr), "v"(goff),
                  "s"((unsigned long long)(size_t)gbase)
               : "memory");
}
__device__ __forceinline__ void wait_async0() {
  asm volatile("s_wait_asynccnt 0x0" ::: "memory");
}
__device__ __forceinline__ uint lds_off(const void* p) {
  return (uint)(size_t)p;  // addrspacecast(3->0): LDS offset lives in addr[31:0]
}

// A fragment: 16(M) x 32(K) bf16, row-major in LDS with stride lda (elements).
// Contiguous k order used here: lanes 0-15 -> M=lane, K=0-15;
//             lanes 16-31 -> M=lane-16, K=16-31.
__device__ __forceinline__ v16bf load_A(const __bf16* base, int lda) {
  const int lane = threadIdx.x & 31;
  const __bf16* p = base + (lane & 15) * lda + ((lane >> 4) << 4);
  v16bf a;
#pragma unroll
  for (int v = 0; v < 8; ++v) {
    const int k = v * 2;
    a[2 * v]     = p[k];
    a[2 * v + 1] = p[k + 1];
  }
  return a;
}

// B fragment: 32(K) x 16(N) bf16 sourced from an N-major LDS image Bt[n][k]
// (stride ldt). Contiguous k order used here: lanes 0-15 K=0-15, lanes 16-31 K=16-31, N=lane&15.
__device__ __forceinline__ v16bf load_B(const __bf16* base, int ldt) {
  const int lane = threadIdx.x & 31;
  const __bf16* p = base + (lane & 15) * ldt + ((lane >> 4) << 4);
  v16bf b;
#pragma unroll
  for (int v = 0; v < 8; ++v) {
    b[2 * v]     = p[2 * v];
    b[2 * v + 1] = p[2 * v + 1];
  }
  return b;
}

__global__ void f2bf(const float* __restrict__ in, __bf16* __restrict__ out, int n) {
  for (int i = blockIdx.x * blockDim.x + threadIdx.x; i < n;
       i += blockDim.x * gridDim.x)
    out[i] = (__bf16)in[i];
}

// ---------------- QKV projection: per (b,h), [S,D] x [D,DH] + bias ------------
__global__ void __launch_bounds__(256)
qkv_gemm(const __bf16* __restrict__ xb,
         const __bf16* __restrict__ Wqb, const __bf16* __restrict__ Wkb,
         const __bf16* __restrict__ Wvb,
         const float* __restrict__ bq, const float* __restrict__ bk,
         const float* __restrict__ bv,
         __bf16* __restrict__ Q, __bf16* __restrict__ K, __bf16* __restrict__ V) {
  __shared__ alignas(16) __bf16 Xs[128 * 40];  // stride 40 elems = 80 B (16B-mult)
  __shared__ __bf16 Ws[64 * 36];
  const int tid = threadIdx.x, lane = tid & 31, wave = tid >> 5;
  const int bh = blockIdx.y, b = bh / Hn, h = bh % Hn;
  const int s0 = blockIdx.x * 128;
  const int which = blockIdx.z;
  const __bf16* W = (which == 0 ? Wqb : which == 1 ? Wkb : Wvb) + (size_t)h * Dm * Dh;
  const float* bias = (which == 0 ? bq : which == 1 ? bk : bv) + h * Dh;
  __bf16* Out = (which == 0 ? Q : which == 1 ? K : V) + (size_t)bh * Sq * Dh;
  const uint ldsXs = lds_off(&Xs[0]);

  v8f acc[4] = {};
  for (int kk = 0; kk < Dm; kk += 32) {
    // X tile 128x32 bf16 via async DMA: 128 rows x 4 chunks of 16B
    const __bf16* xg = xb + ((size_t)b * Sq + s0) * Dm + kk;
#pragma unroll
    for (int it = 0; it < 2; ++it) {
      const int c = tid + it * 256;
      const int row = c >> 2, cr = c & 3;
      async_ld128(ldsXs + row * 80 + cr * 16, row * (Dm * 2) + cr * 16, xg);
    }
    for (int e = tid; e < 64 * 32; e += 256) {  // W tile transposed -> N-major
      const int n = e >> 5, k = e & 31;
      Ws[n * 36 + k] = W[(size_t)(kk + k) * Dh + n];
    }
    wait_async0();
    __syncthreads();
    const v16bf a = load_A(Xs + wave * 16 * 40, 40);
#pragma unroll
    for (int i = 0; i < 4; ++i) {
      const v16bf bfr = load_B(Ws + i * 16 * 36, 36);
      acc[i] = wmma_bf16(a, bfr, acc[i]);
    }
    __syncthreads();
  }
  const int half = (lane >> 4) << 3;
#pragma unroll
  for (int i = 0; i < 4; ++i) {
    const int dh = i * 16 + (lane & 15);
    const float bia = bias[dh];
#pragma unroll
    for (int r = 0; r < 8; ++r) {
      const int srow = s0 + wave * 16 + half + r;
      Out[(size_t)srow * Dh + dh] = (__bf16)(acc[i][r] + bia);
    }
  }
}

// ---------------- Flash attention: block = (b,h, 64 q-rows), 4 waves ---------
__global__ void __launch_bounds__(128)
flash_attn(const __bf16* __restrict__ Q, const __bf16* __restrict__ K,
           const __bf16* __restrict__ V, __bf16* __restrict__ ctx) {
  __shared__ alignas(16) __bf16 Qs[64 * 72];  // 144 B rows (16B-mult)
  __shared__ alignas(16) __bf16 Ks[64 * 72];  // K rows row-major == N-major B
  __shared__ __bf16 Vt[64 * 72];              // V transposed -> N-major B image
  __shared__ float  Ss[4 * 16 * 68];          // per-wave score scratch
  __shared__ __bf16 Pt[4 * 16 * 72];          // per-wave P tile (A operand)
  const int tid = threadIdx.x, lane = tid & 31, wave = tid >> 5;
  const int bh = blockIdx.y, b = bh / Hn, h = bh % Hn;
  const int q0 = blockIdx.x * 64;
  const __bf16* Qg = Q + ((size_t)bh * Sq + q0) * Dh;
  const __bf16* Kg0 = K + (size_t)bh * Sq * Dh;
  const __bf16* Vg0 = V + (size_t)bh * Sq * Dh;
  const uint ldsQs = lds_off(&Qs[0]);
  const uint ldsKs = lds_off(&Ks[0]);

  // Q tile 64x64 bf16 = 64 rows x 8 chunks of 16B, async DMA
#pragma unroll
  for (int it = 0; it < 4; ++it) {
    const int c = tid + it * 128;
    const int row = c >> 3, cr = c & 7;
    async_ld128(ldsQs + row * 144 + cr * 16, row * 128 + cr * 16, Qg);
  }
  v8f O[4] = {};
  float m_i = -1e30f, l_i = 0.f, alpha = 1.f;
  float* Sw = Ss + wave * 16 * 68;
  __bf16* Pw = Pt + wave * 16 * 72;
  const int half = (lane >> 4) << 3;
  const float scale = 0.125f;  // 1/sqrt(DH)

  for (int kv0 = 0; kv0 < Sq; kv0 += 64) {
    const __bf16* kg = Kg0 + (size_t)kv0 * Dh;
#pragma unroll
    for (int it = 0; it < 4; ++it) {
      const int c = tid + it * 128;
      const int row = c >> 3, cr = c & 7;
      async_ld128(ldsKs + row * 144 + cr * 16, row * 128 + cr * 16, kg);
    }
    const __bf16* vg = Vg0 + (size_t)kv0 * Dh;
    for (int e = tid; e < 64 * 64; e += 128) {
      const int kv = e >> 6, dh = e & 63;
      Vt[dh * 72 + kv] = vg[kv * 64 + dh];
    }
    wait_async0();
    __syncthreads();
    // scores S = Q K^T (16 x 64 per wave)
    v8f sacc[4] = {};
#pragma unroll
    for (int ks = 0; ks < 2; ++ks) {
      const v16bf a = load_A(Qs + wave * 16 * 72 + ks * 32, 72);
#pragma unroll
      for (int i = 0; i < 4; ++i) {
        const v16bf bfr = load_B(Ks + i * 16 * 72 + ks * 32, 72);
        sacc[i] = wmma_bf16(a, bfr, sacc[i]);
      }
    }
#pragma unroll
    for (int i = 0; i < 4; ++i)
#pragma unroll
      for (int r = 0; r < 8; ++r)
        Sw[(half + r) * 68 + i * 16 + (lane & 15)] = sacc[i][r] * scale;
    __syncthreads();
    if (lane < 16) {  // lane r owns q-row r of this wave
      float* srow = Sw + lane * 68;
      float mt = -1e30f;
      for (int j = 0; j < 64; ++j) mt = fmaxf(mt, srow[j]);
      const float mnew = fmaxf(m_i, mt);
      alpha = __expf(m_i - mnew);
      float ls = 0.f;
      for (int j = 0; j < 64; ++j) {
        const float p = __expf(srow[j] - mnew);
        ls += p;
        Pw[lane * 72 + j] = (__bf16)p;
      }
      l_i = alpha * l_i + ls;
      m_i = mnew;
    }
    __syncthreads();
    float asc[8];
#pragma unroll
    for (int r = 0; r < 8; ++r) asc[r] = __shfl(alpha, half + r, 32);
#pragma unroll
    for (int i = 0; i < 4; ++i)
#pragma unroll
      for (int r = 0; r < 8; ++r) O[i][r] *= asc[r];
    // O += P V
#pragma unroll
    for (int ks = 0; ks < 2; ++ks) {
      const v16bf a = load_A(Pw + ks * 32, 72);
#pragma unroll
      for (int i = 0; i < 4; ++i) {
        const v16bf bfr = load_B(Vt + i * 16 * 72 + ks * 32, 72);
        O[i] = wmma_bf16(a, bfr, O[i]);
      }
    }
    __syncthreads();
  }
  const float invl = (lane < 16 && l_i > 0.f) ? 1.f / l_i : 1.f;
  float isc[8];
#pragma unroll
  for (int r = 0; r < 8; ++r) isc[r] = __shfl(invl, half + r, 32);
#pragma unroll
  for (int i = 0; i < 4; ++i) {
    const int dh = i * 16 + (lane & 15);
#pragma unroll
    for (int r = 0; r < 8; ++r) {
      const int s = q0 + wave * 16 + half + r;
      ctx[((size_t)b * Sq + s) * Dm + h * Dh + dh] = (__bf16)(O[i][r] * isc[r]);
    }
  }
}

// ---------------- Output projection: [B*S, D] x [D, D] + bo, f32 out --------
__global__ void __launch_bounds__(256)
out_proj(const __bf16* __restrict__ ctx, const __bf16* __restrict__ Wob,
         const float* __restrict__ bo, float* __restrict__ out) {
  __shared__ alignas(16) __bf16 Cs[128 * 40];
  __shared__ __bf16 Ws[64 * 36];
  const int tid = threadIdx.x, lane = tid & 31, wave = tid >> 5;
  const int r0 = blockIdx.x * 128;
  const int n0 = blockIdx.y * 64;
  const uint ldsCs = lds_off(&Cs[0]);
  v8f acc[4] = {};
  for (int kk = 0; kk < Dm; kk += 32) {
    const __bf16* cg = ctx + (size_t)r0 * Dm + kk;
#pragma unroll
    for (int it = 0; it < 2; ++it) {
      const int c = tid + it * 256;
      const int row = c >> 2, cr = c & 3;
      async_ld128(ldsCs + row * 80 + cr * 16, row * (Dm * 2) + cr * 16, cg);
    }
    for (int e = tid; e < 64 * 32; e += 256) {
      const int n = e >> 5, k = e & 31;
      Ws[n * 36 + k] = Wob[(size_t)(kk + k) * Dm + n0 + n];
    }
    wait_async0();
    __syncthreads();
    const v16bf a = load_A(Cs + wave * 16 * 40, 40);
#pragma unroll
    for (int i = 0; i < 4; ++i) {
      const v16bf bfr = load_B(Ws + i * 16 * 36, 36);
      acc[i] = wmma_bf16(a, bfr, acc[i]);
    }
    __syncthreads();
  }
  const int half = (lane >> 4) << 3;
#pragma unroll
  for (int i = 0; i < 4; ++i) {
    const int n = n0 + i * 16 + (lane & 15);
    const float bb = bo[n];
#pragma unroll
    for (int r = 0; r < 8; ++r) {
      const int row = r0 + wave * 16 + half + r;
      out[(size_t)row * Dm + n] = acc[i][r] + bb;
    }
  }
}

extern "C" void kernel_launch(void* const* d_in, const int* in_sizes, int n_in,
                              void* d_out, int out_size, void* d_ws, size_t ws_size,
                              hipStream_t stream) {
  (void)in_sizes; (void)n_in; (void)out_size; (void)ws_size;
  const float* x  = (const float*)d_in[0];
  const float* Wq = (const float*)d_in[1];
  const float* bq = (const float*)d_in[2];
  const float* Wk = (const float*)d_in[3];
  const float* bk = (const float*)d_in[4];
  const float* Wv = (const float*)d_in[5];
  const float* bv = (const float*)d_in[6];
  const float* Wo = (const float*)d_in[7];
  const float* bo = (const float*)d_in[8];
  float* out = (float*)d_out;

  char* ws = (char*)d_ws;
  size_t off = 0;
  auto alloc = [&](size_t bytes) {
    char* p = ws + off;
    off += (bytes + 255) & ~(size_t)255;
    return p;
  };
  const size_t nX = (size_t)Bb * Sq * Dm;        // 4,194,304
  const size_t nW = (size_t)Hn * Dm * Dh;        // 1,048,576
  __bf16* xb  = (__bf16*)alloc(nX * 2);
  __bf16* Wqb = (__bf16*)alloc(nW * 2);
  __bf16* Wkb = (__bf16*)alloc(nW * 2);
  __bf16* Wvb = (__bf16*)alloc(nW * 2);
  __bf16* Wob = (__bf16*)alloc(nW * 2);
  __bf16* Qb  = (__bf16*)alloc(nX * 2);
  __bf16* Kb  = (__bf16*)alloc(nX * 2);
  __bf16* Vb  = (__bf16*)alloc(nX * 2);
  __bf16* ctxb = (__bf16*)alloc(nX * 2);

  f2bf<<<2048, 256, 0, stream>>>(x,  xb,  (int)nX);
  f2bf<<<1024, 256, 0, stream>>>(Wq, Wqb, (int)nW);
  f2bf<<<1024, 256, 0, stream>>>(Wk, Wkb, (int)nW);
  f2bf<<<1024, 256, 0, stream>>>(Wv, Wvb, (int)nW);
  f2bf<<<1024, 256, 0, stream>>>(Wo, Wob, (int)nW);

  dim3 gq(Sq / 128, Bb * Hn, 3);
  qkv_gemm<<<gq, 256, 0, stream>>>(xb, Wqb, Wkb, Wvb, bq, bk, bv, Qb, Kb, Vb);

  dim3 ga(Sq / 64, Bb * Hn);
  flash_attn<<<ga, 128, 0, stream>>>(Qb, Kb, Vb, ctxb);

  dim3 go((Bb * Sq) / 128, Dm / 64);
  out_proj<<<go, 256, 0, stream>>>(ctxb, Wob, bo, out);
}
